// MTMD_Domain_Task_Expert_Wo_Gate_Base_6227702579368
// MI455X (gfx1250) — matrix operand order repaired and checked by reading, not compiled, not run on hardware
//
#include <hip/hip_runtime.h>
#include <cstdint>
#include <cstddef>

// ---------------- types ----------------
typedef __bf16 bf16_t;
typedef bf16_t bf16x8  __attribute__((ext_vector_type(8)));
typedef bf16_t bf16x16 __attribute__((ext_vector_type(16)));
typedef float  f32x8   __attribute__((ext_vector_type(8)));

#define BATCH    65536
#define N1       1024      // skip(256) + 3 * folded-STAR(256)
#define N2       768       // shared(128) + 3*dom(128) + 2*task(128)
#define BN_SCALE 0.9999950000374996f

static __device__ __forceinline__ unsigned short f2bf(float f) {
  unsigned u = __builtin_bit_cast(unsigned, f);
  u += 0x7FFFu + ((u >> 16) & 1u);
  return (unsigned short)(u >> 16);
}
static __device__ __forceinline__ bf16x16 cat8(bf16x8 a, bf16x8 b) {
  return __builtin_shufflevector(a, b, 0,1,2,3,4,5,6,7,8,9,10,11,12,13,14,15);
}
// WMMA A frag (16x32 bf16): lane holds row=lane&15; K octets {kb+hi*8, kb+16+hi*8}
static __device__ __forceinline__ bf16x16 frag_a(const unsigned short* rowp, int kb, int hi) {
  bf16x8 a0 = *reinterpret_cast<const bf16x8*>(rowp + kb + hi * 8);
  bf16x8 a1 = *reinterpret_cast<const bf16x8*>(rowp + kb + 16 + hi * 8);
  return cat8(a0, a1);
}
// WMMA B frag (32x16 bf16) from transposed [N,K] weights: lane holds col=lane&15,
// K octets {kb+hi*8, kb+16+hi*8}
static __device__ __forceinline__ bf16x16 frag_b(const unsigned short* wrow, int kb, int hi) {
  bf16x8 b0 = *reinterpret_cast<const bf16x8*>(wrow + kb + hi * 8);
  bf16x8 b1 = *reinterpret_cast<const bf16x8*>(wrow + kb + 16 + hi * 8);
  return cat8(b0, b1);
}
static __device__ __forceinline__ float wredsum(float v) {
  v += __shfl_xor(v, 1, 32);  v += __shfl_xor(v, 2, 32);
  v += __shfl_xor(v, 4, 32);  v += __shfl_xor(v, 8, 32);
  v += __shfl_xor(v, 16, 32); return v;
}

// ---------------- prep kernels ----------------
__global__ __launch_bounds__(256) void k_prep_skipT(const float* __restrict__ skip_W,
                                                    unsigned short* __restrict__ WB1T) {
  int idx = blockIdx.x * 256 + threadIdx.x;            // 256*512
  int n = idx >> 9, k = idx & 511;
  WB1T[(size_t)n * 512 + k] = f2bf(skip_W[(size_t)k * 256 + n]);
}

// rows 256+d*256+o of WB1T: C_d[k][o] = sum_h slot_W[d,k,h]*shared_W[k,h]*star_W[h,o]
__global__ __launch_bounds__(256) void k_prep_star(const float* __restrict__ slot_W,
                                                   const float* __restrict__ shared_W,
                                                   const float* __restrict__ star_W,
                                                   unsigned short* __restrict__ WB1T) {
  __shared__ float sw[512];
  int d = blockIdx.x >> 8;          // 768 blocks: d in [0,3), o in [0,256)
  int o = blockIdx.x & 255;
  int tid = threadIdx.x;
  sw[tid]       = star_W[(size_t)tid * 256 + o];
  sw[tid + 256] = star_W[(size_t)(tid + 256) * 256 + o];
  __syncthreads();
  for (int kk = 0; kk < 2; ++kk) {
    int k = tid + kk * 256;
    const float* sl = slot_W + ((size_t)d * 512 + k) * 512;
    const float* sh = shared_W + (size_t)k * 512;
    float acc = 0.f;
#pragma unroll 4
    for (int h = 0; h < 512; ++h) acc += sl[h] * sh[h] * sw[h];
    WB1T[(size_t)(256 + d * 256 + o) * 512 + k] = f2bf(acc);
  }
}

__global__ __launch_bounds__(256) void k_prep_cbias(const float* __restrict__ slot_b,
                                                    const float* __restrict__ shared_b,
                                                    const float* __restrict__ star_W,
                                                    const float* __restrict__ star_b,
                                                    float* __restrict__ cbias) {
  int d = blockIdx.x, o = threadIdx.x;
  float acc = star_b[o];
  for (int h = 0; h < 512; ++h)
    acc += (slot_b[d * 512 + h] + shared_b[h]) * star_W[(size_t)h * 256 + o];
  cbias[d * 256 + o] = acc;
}

__global__ __launch_bounds__(256) void k_prep_wb2(const float* __restrict__ sexp_W,
                                                  const float* __restrict__ dexp_W,
                                                  const float* __restrict__ texp_W,
                                                  unsigned short* __restrict__ WB2T) {
  int idx = blockIdx.x * 256 + threadIdx.x;            // 768*256
  int n = idx >> 8, k = idx & 255;
  float v;
  if (n < 128)      v = sexp_W[(size_t)k * 128 + n];
  else if (n < 512) { int d = (n - 128) >> 7, o = (n - 128) & 127;
                      v = dexp_W[((size_t)d * 256 + k) * 128 + o]; }
  else              { int t = (n - 512) >> 7, o = (n - 512) & 127;
                      v = texp_W[((size_t)t * 256 + k) * 128 + o]; }
  WB2T[(size_t)n * 256 + k] = f2bf(v);
}

__global__ __launch_bounds__(256) void k_prep_wt1(const float* __restrict__ tw_W1,
                                                  unsigned short* __restrict__ WT1T) {
  int idx = blockIdx.x * 256 + threadIdx.x;            // 6*128*384
  int j = idx / (128 * 384);
  int rem = idx - j * 128 * 384;
  int n = rem / 384, k = rem - n * 384;
  WT1T[((size_t)j * 128 + n) * 384 + k] = f2bf(tw_W1[((size_t)j * 384 + k) * 128 + n]);
}

// ---------------- embedding gather -> X bf16 [B,512] ----------------
__global__ __launch_bounds__(256) void k_embed(const int* __restrict__ x,
                                               const float* __restrict__ tabs,
                                               unsigned short* __restrict__ X) {
  int tid = threadIdx.x;
  int s = blockIdx.x * 16 + (tid >> 4);
  int f = tid & 15;
  int idx = x[(size_t)f * BATCH + s];
  const float* src = tabs + ((size_t)f * 100000 + idx) * 32;
  unsigned short* dst = X + (size_t)s * 512 + f * 32;
#pragma unroll
  for (int e = 0; e < 32; e += 4) {
    float4 v = *reinterpret_cast<const float4*>(src + e);
    uint2 w;
    w.x = (unsigned)f2bf(v.x) | ((unsigned)f2bf(v.y) << 16);
    w.y = (unsigned)f2bf(v.z) | ((unsigned)f2bf(v.w) << 16);
    *reinterpret_cast<uint2*>(dst + e) = w;
  }
}

// ---------------- GEMM1: X[B,512] x WB1T[1024,512]^T, fused LN/gather -> H bf16 ----------------
__global__ __launch_bounds__(256) void k_gemm1(const unsigned short* __restrict__ X,
                                               const unsigned short* __restrict__ WB1T,
                                               const float* __restrict__ cbias,
                                               const float* __restrict__ skip_b,
                                               const float* __restrict__ skip_g,
                                               const float* __restrict__ skip_beta,
                                               const float* __restrict__ star_g,
                                               const float* __restrict__ star_beta,
                                               const int* __restrict__ domain_id,
                                               unsigned short* __restrict__ H) {
  __shared__ __align__(16) unsigned short sA[16 * 512];   // 16 KB A tile
  __shared__ float sOut[16 * N1];                         // 64 KB
  int tid = threadIdx.x, wave = tid >> 5, lane = tid & 31;
  int row = lane & 15, hi = lane >> 4;
  int m0 = blockIdx.x * 16;
  int nbase = wave * 128;
  // cooperative stage of contiguous 16x512 A tile
  {
    const bf16x8* src = reinterpret_cast<const bf16x8*>(X + (size_t)m0 * 512);
    bf16x8* dst = reinterpret_cast<bf16x8*>(sA);
#pragma unroll
    for (int i = 0; i < 4; ++i) dst[tid + i * 256] = src[tid + i * 256];
  }
  __syncthreads();
  f32x8 acc[8] = {};
  const unsigned short* arow = sA + row * 512;
  for (int kb = 0; kb < 512; kb += 32) {
    bf16x16 a = frag_a(arow, kb, hi);
    bf16x16 bfr[8];
#pragma unroll
    for (int t = 0; t < 8; ++t)
      bfr[t] = frag_b(WB1T + (size_t)(nbase + t * 16 + row) * 512, kb, hi);
#pragma unroll
    for (int t = 0; t < 8; ++t)
      acc[t] = __builtin_amdgcn_wmma_f32_16x16x32_bf16(false, a, false, bfr[t],
                                                       (short)0, acc[t], false, false);
  }
#pragma unroll
  for (int t = 0; t < 8; ++t)
#pragma unroll
    for (int v = 0; v < 8; ++v)
      sOut[(v + 8 * hi) * N1 + nbase + t * 16 + row] = acc[t][v];
  __syncthreads();
  // epilogue: wave handles 2 rows; LN(skip) + domain-gathered LN(star) + add
  for (int rr = wave * 2; rr < wave * 2 + 2; ++rr) {
    int bidx = m0 + rr;
    int d = domain_id[bidx];
    const float* srow = sOut + rr * N1;
    const float* cb = cbias + d * 256;
    float sk[8], hv[8];
    float s = 0.f, s2 = 0.f;
#pragma unroll
    for (int i = 0; i < 8; ++i) { int c = lane + 32 * i;
      float xv = srow[c] + skip_b[c]; sk[i] = xv; s += xv; s2 += xv * xv; }
    s = wredsum(s); s2 = wredsum(s2);
    float mu = s * (1.f / 256.f), var = s2 * (1.f / 256.f) - mu * mu;
    float rstd = rsqrtf(var + 1e-5f);
#pragma unroll
    for (int i = 0; i < 8; ++i) { int c = lane + 32 * i;
      sk[i] = fmaxf((sk[i] - mu) * rstd * skip_g[c] + skip_beta[c], 0.f); }
    s = 0.f; s2 = 0.f;
#pragma unroll
    for (int i = 0; i < 8; ++i) { int c = lane + 32 * i;
      float xv = srow[256 + d * 256 + c] + cb[c]; hv[i] = xv; s += xv; s2 += xv * xv; }
    s = wredsum(s); s2 = wredsum(s2);
    mu = s * (1.f / 256.f); var = s2 * (1.f / 256.f) - mu * mu;
    rstd = rsqrtf(var + 1e-5f);
#pragma unroll
    for (int i = 0; i < 8; ++i) { int c = lane + 32 * i;
      float h = fmaxf((hv[i] - mu) * rstd * star_g[c] + star_beta[c], 0.f) + sk[i];
      H[(size_t)bidx * 256 + c] = f2bf(h); }
  }
}

// ---------------- GEMM2: H[B,256] x WB2T[768,256]^T, 6 group LNs -> FEA bf16 ----------------
__global__ __launch_bounds__(192) void k_gemm2(const unsigned short* __restrict__ H,
                                               const unsigned short* __restrict__ WB2T,
                                               const float* __restrict__ sexp_b,
                                               const float* __restrict__ sexp_g,
                                               const float* __restrict__ sexp_beta,
                                               const float* __restrict__ dexp_b,
                                               const float* __restrict__ dexp_g,
                                               const float* __restrict__ dexp_beta,
                                               const float* __restrict__ texp_b,
                                               const float* __restrict__ texp_g,
                                               const float* __restrict__ texp_beta,
                                               unsigned short* __restrict__ FEA) {
  __shared__ __align__(16) unsigned short sA[16 * 256];   // 8 KB A tile
  __shared__ float sOut[16 * N2];                         // 48 KB
  int tid = threadIdx.x, wave = tid >> 5, lane = tid & 31;
  int row = lane & 15, hi = lane >> 4;
  int m0 = blockIdx.x * 16;
  int nbase = wave * 128;
  {
    const bf16x8* src = reinterpret_cast<const bf16x8*>(H + (size_t)m0 * 256);
    bf16x8* dst = reinterpret_cast<bf16x8*>(sA);
    for (int i = tid; i < 512; i += 192) dst[i] = src[i];
  }
  __syncthreads();
  f32x8 acc[8] = {};
  const unsigned short* arow = sA + row * 256;
  for (int kb = 0; kb < 256; kb += 32) {
    bf16x16 a = frag_a(arow, kb, hi);
    bf16x16 bfr[8];
#pragma unroll
    for (int t = 0; t < 8; ++t)
      bfr[t] = frag_b(WB2T + (size_t)(nbase + t * 16 + row) * 256, kb, hi);
#pragma unroll
    for (int t = 0; t < 8; ++t)
      acc[t] = __builtin_amdgcn_wmma_f32_16x16x32_bf16(false, a, false, bfr[t],
                                                       (short)0, acc[t], false, false);
  }
#pragma unroll
  for (int t = 0; t < 8; ++t)
#pragma unroll
    for (int v = 0; v < 8; ++v)
      sOut[(v + 8 * hi) * N2 + nbase + t * 16 + row] = acc[t][v];
  __syncthreads();
  for (int rr = wave; rr < 16; rr += 6) {
    int bidx = m0 + rr;
    const float* srow = sOut + rr * N2;
    for (int g = 0; g < 6; ++g) {
      const float *bias, *gain, *beta;
      if (g == 0)     { bias = sexp_b;                 gain = sexp_g;                 beta = sexp_beta; }
      else if (g < 4) { bias = dexp_b + (g - 1) * 128; gain = dexp_g + (g - 1) * 128; beta = dexp_beta + (g - 1) * 128; }
      else            { bias = texp_b + (g - 4) * 128; gain = texp_g + (g - 4) * 128; beta = texp_beta + (g - 4) * 128; }
      float v[4], s = 0.f, s2 = 0.f;
#pragma unroll
      for (int i = 0; i < 4; ++i) { int cl = lane + 32 * i;
        float xv = srow[g * 128 + cl] + bias[cl]; v[i] = xv; s += xv; s2 += xv * xv; }
      s = wredsum(s); s2 = wredsum(s2);
      float mu = s * (1.f / 128.f), var = s2 * (1.f / 128.f) - mu * mu;
      float rstd = rsqrtf(var + 1e-5f);
#pragma unroll
      for (int i = 0; i < 4; ++i) { int cl = lane + 32 * i;
        float hv = fmaxf((v[i] - mu) * rstd * gain[cl] + beta[cl], 0.f);
        FEA[(size_t)bidx * N2 + g * 128 + cl] = f2bf(hv); }
    }
  }
}

// ---------------- GEMM3: one block stages a FEA tile, loops all 6 towers ----------------
__global__ __launch_bounds__(128) void k_gemm3(const unsigned short* __restrict__ FEA,
                                               const unsigned short* __restrict__ WT1T,
                                               const float* __restrict__ tw_b1,
                                               const float* __restrict__ tw_g,
                                               const float* __restrict__ tw_beta,
                                               const float* __restrict__ tw_W2,
                                               const float* __restrict__ tw_b2,
                                               const int* __restrict__ domain_id,
                                               float* __restrict__ out) {
  __shared__ __align__(16) unsigned short sFEA[16 * N2];  // 24 KB
  __shared__ float sOut[16 * 128];                        // 8 KB
  int tid = threadIdx.x, wave = tid >> 5, lane = tid & 31;
  int row = lane & 15, hi = lane >> 4;
  int m0 = blockIdx.x * 16;
  int nbase = wave * 32;
  {
    const bf16x8* src = reinterpret_cast<const bf16x8*>(FEA + (size_t)m0 * N2);
    bf16x8* dst = reinterpret_cast<bf16x8*>(sFEA);
#pragma unroll
    for (int i = 0; i < 12; ++i) dst[tid + i * 128] = src[tid + i * 128];
  }
  __syncthreads();
  int r = tid >> 3, sub = tid & 7;          // epilogue mapping (8 threads / row)
  for (int j = 0; j < 6; ++j) {
    int dj = j % 3, tj = j % 2;
    f32x8 acc[2] = {};
    const unsigned short* arow = sFEA + row * N2;
    for (int kb = 0; kb < 384; kb += 32) {
      int blk = kb >> 7;
      int base = (blk == 0) ? 0 : ((blk == 1) ? (128 + dj * 128) : (512 + tj * 128));
      bf16x16 a = frag_a(arow + base, kb & 127, hi);
      bf16x16 bfr[2];
#pragma unroll
      for (int t = 0; t < 2; ++t)
        bfr[t] = frag_b(WT1T + ((size_t)j * 128 + nbase + t * 16 + row) * 384, kb, hi);
#pragma unroll
      for (int t = 0; t < 2; ++t)
        acc[t] = __builtin_amdgcn_wmma_f32_16x16x32_bf16(false, a, false, bfr[t],
                                                         (short)0, acc[t], false, false);
    }
    __syncthreads();   // previous epilogue's sOut reads are done
#pragma unroll
    for (int t = 0; t < 2; ++t)
#pragma unroll
      for (int v = 0; v < 8; ++v)
        sOut[(v + 8 * hi) * 128 + nbase + t * 16 + row] = acc[t][v];
    __syncthreads();
    // eval-BN + ReLU + dot(tw_W2) + sigmoid, selective write
    const float* b1 = tw_b1 + j * 128;
    const float* gg = tw_g + j * 128;
    const float* bt = tw_beta + j * 128;
    const float* w2 = tw_W2 + j * 128;
    float s = 0.f;
#pragma unroll
    for (int i = 0; i < 16; ++i) {
      int c = sub * 16 + i;
      float v = (sOut[r * 128 + c] + b1[c]) * (gg[c] * BN_SCALE) + bt[c];
      s += fmaxf(v, 0.f) * w2[c];
    }
    s += __shfl_xor(s, 1, 32); s += __shfl_xor(s, 2, 32); s += __shfl_xor(s, 4, 32);
    if (sub == 0) {
      int bidx = m0 + r;
      if (domain_id[bidx] == dj) {
        float logit = s + tw_b2[j];
        out[(size_t)bidx * 2 + (j / 3)] = 1.f / (1.f + expf(-logit));
      }
    }
  }
}

// ---------------- host launch ----------------
extern "C" void kernel_launch(void* const* d_in, const int* in_sizes, int n_in,
                              void* d_out, int out_size, void* d_ws, size_t ws_size,
                              hipStream_t stream) {
  (void)in_sizes; (void)n_in; (void)out_size; (void)ws_size;
  const int*   x          = (const int*)  d_in[0];
  const int*   domain_id  = (const int*)  d_in[1];
  const float* emb_tables = (const float*)d_in[2];
  const float* skip_W     = (const float*)d_in[3];
  const float* skip_b     = (const float*)d_in[4];
  const float* skip_g     = (const float*)d_in[5];
  const float* skip_beta  = (const float*)d_in[6];
  const float* shared_W   = (const float*)d_in[7];
  const float* shared_b   = (const float*)d_in[8];
  const float* slot_W     = (const float*)d_in[9];
  const float* slot_b     = (const float*)d_in[10];
  const float* star_W     = (const float*)d_in[11];
  const float* star_b     = (const float*)d_in[12];
  const float* star_g     = (const float*)d_in[13];
  const float* star_beta  = (const float*)d_in[14];
  const float* sexp_W     = (const float*)d_in[15];
  const float* sexp_b     = (const float*)d_in[16];
  const float* sexp_g     = (const float*)d_in[17];
  const float* sexp_beta  = (const float*)d_in[18];
  const float* dexp_W     = (const float*)d_in[19];
  const float* dexp_b     = (const float*)d_in[20];
  const float* dexp_g     = (const float*)d_in[21];
  const float* dexp_beta  = (const float*)d_in[22];
  const float* texp_W     = (const float*)d_in[23];
  const float* texp_b     = (const float*)d_in[24];
  const float* texp_g     = (const float*)d_in[25];
  const float* texp_beta  = (const float*)d_in[26];
  const float* tw_W1      = (const float*)d_in[27];
  const float* tw_b1      = (const float*)d_in[28];
  const float* tw_g       = (const float*)d_in[29];
  const float* tw_beta    = (const float*)d_in[30];
  const float* tw_W2      = (const float*)d_in[31];
  const float* tw_b2      = (const float*)d_in[32];
  float* out = (float*)d_out;

  char* ws = (char*)d_ws;
  unsigned short* X    = (unsigned short*)(ws + (size_t)0);           // 64 MB
  unsigned short* H    = (unsigned short*)(ws + (size_t)67108864);    // 32 MB
  unsigned short* FEA  = (unsigned short*)(ws + (size_t)100663296);   // 96 MB
  unsigned short* WB1T = (unsigned short*)(ws + (size_t)201326592);   // 1 MB
  unsigned short* WB2T = (unsigned short*)(ws + (size_t)202375168);   // 384 KB
  unsigned short* WT1T = (unsigned short*)(ws + (size_t)202768384);   // 576 KB
  float*          CBIAS = (float*)        (ws + (size_t)203358208);   // 3 KB

  // weight prep (independent, serialized on stream)
  k_prep_skipT<<<512, 256, 0, stream>>>(skip_W, WB1T);
  k_prep_star <<<768, 256, 0, stream>>>(slot_W, shared_W, star_W, WB1T);
  k_prep_cbias<<<3, 256, 0, stream>>>(slot_b, shared_b, star_W, star_b, CBIAS);
  k_prep_wb2  <<<768, 256, 0, stream>>>(sexp_W, dexp_W, texp_W, WB2T);
  k_prep_wt1  <<<1152, 256, 0, stream>>>(tw_W1, WT1T);

  // activation pipeline
  k_embed<<<BATCH / 16, 256, 0, stream>>>(x, emb_tables, X);
  k_gemm1<<<BATCH / 16, 256, 0, stream>>>(X, WB1T, CBIAS, skip_b, skip_g, skip_beta,
                                          star_g, star_beta, domain_id, H);
  k_gemm2<<<BATCH / 16, 192, 0, stream>>>(H, WB2T, sexp_b, sexp_g, sexp_beta,
                                          dexp_b, dexp_g, dexp_beta,
                                          texp_b, texp_g, texp_beta, FEA);
  k_gemm3<<<BATCH / 16, 128, 0, stream>>>(FEA, WT1T, tw_b1, tw_g, tw_beta,
                                          tw_W2, tw_b2, domain_id, out);
}
